// Nerf_74448963109082
// MI455X (gfx1250) — matrix operand order repaired and checked by reading, not compiled, not run on hardware
//
#include <hip/hip_runtime.h>

typedef __attribute__((ext_vector_type(16))) _Float16 v16h;
typedef __attribute__((ext_vector_type(8)))  _Float16 v8h;
typedef __attribute__((ext_vector_type(8)))  float    v8f;

#define TPB 128          // 4 wave32
#define FS  72           // row stride in halves (multiple of 8 -> 16B aligned chunks)

// ---- hash-grid encode for one level, one point per lane ----
template<int HS, int D>
__device__ __forceinline__ void encode_level(const float* __restrict__ tab,
                                             float p0, float p1, float p2,
                                             _Float16* __restrict__ dst)
{
    float f0 = floorf(p0), f1 = floorf(p1), f2 = floorf(p2);
    int   fi0 = (int)f0, fi1 = (int)f1, fi2 = (int)f2;
    int   ci0 = (int)ceilf(p0), ci1 = (int)ceilf(p1), ci2 = (int)ceilf(p2);
    float r0 = p0 - f0, r1 = p1 - f1, r2 = p2 - f2;

    float acc[D];
#pragma unroll
    for (int d = 0; d < D; ++d) acc[d] = 0.0f;

#pragma unroll
    for (int c = 0; c < 8; ++c) {
        const int i = (c >> 2) & 1, j = (c >> 1) & 1, k = c & 1;
        uint32_t hx = (uint32_t)(i ? ci0 : fi0) * 73856093u;
        uint32_t hy = (uint32_t)(j ? ci1 : fi1) * 19349663u;
        uint32_t hz = (uint32_t)(k ? ci2 : fi2) * 83492791u;
        int32_t  v  = (int32_t)(hx ^ hy ^ hz);
        int32_t  h  = v % HS; if (h < 0) h += HS;           // jnp.mod: non-negative
        // reference quirk: r weights the FLOOR-corner group
        float w = (i ? (1.0f - r0) : r0) * (j ? (1.0f - r1) : r1) * (k ? (1.0f - r2) : r2);
        const float* e = tab + (size_t)h * D;
#pragma unroll
        for (int d = 0; d < D; ++d) acc[d] = fmaf(w, e[d], acc[d]);
    }
#pragma unroll
    for (int d = 0; d < D; ++d) dst[d] = (_Float16)acc[d];
}

__launch_bounds__(TPB)
__global__ void nerf_hashmlp_kernel(const float* __restrict__ x,
                                    const float* __restrict__ e0, const float* __restrict__ e1,
                                    const float* __restrict__ e2, const float* __restrict__ e3,
                                    const float* __restrict__ e4,
                                    const float* __restrict__ Wd1, const float* __restrict__ bd1,
                                    const float* __restrict__ Wd2, const float* __restrict__ bd2,
                                    const float* __restrict__ Wc1, const float* __restrict__ bc1,
                                    const float* __restrict__ Wc2, const float* __restrict__ bc2,
                                    float* __restrict__ out, int n)
{
    __shared__ __attribute__((aligned(16))) _Float16 featL[TPB * FS]; // f16 features, K padded to 64
    __shared__ __attribute__((aligned(16))) _Float16 hidL [TPB * FS]; // f16 hidden activations
    __shared__ __attribute__((aligned(16))) _Float16 w1L  [64 * FS];  // W1^T padded 64x64 f16
    __shared__ float b1L[64];
    __shared__ float w2L[64 * 3];

    const int tid  = threadIdx.x;
    const int lane = tid & 31;
    const int wave = tid >> 5;
    const int g    = blockIdx.x * TPB + tid;

    // -------- Phase A: per-lane hash-grid features --------
    float px = 0.f, py = 0.f, pz = 0.f;
    if (g < n) { px = x[3*g]; py = x[3*g+1]; pz = x[3*g+2]; }

    _Float16* frow = &featL[tid * FS];
    encode_level< 50, 32>(e0, px * 1.0f,  py * 1.0f,  pz * 1.0f,  frow + 0);
    encode_level<200, 16>(e1, px * 2.0f,  py * 2.0f,  pz * 2.0f,  frow + 32);
    encode_level<400,  8>(e2, px * 4.0f,  py * 4.0f,  pz * 4.0f,  frow + 48);
    encode_level<400,  4>(e3, px * 8.0f,  py * 8.0f,  pz * 8.0f,  frow + 56);
    encode_level<400,  3>(e4, px * 20.0f, py * 20.0f, pz * 20.0f, frow + 60);
    frow[63] = (_Float16)0.0f;   // K pad
    __syncthreads();

    // -------- Two passes: dense MLP then color MLP --------
    for (int pass = 0; pass < 2; ++pass) {
        const float* W1p = (pass == 0) ? Wd1 : Wc1;
        const float* b1p = (pass == 0) ? bd1 : bc1;

        // stage W1^T (padded 64x64) in LDS as f16
        for (int idx = tid; idx < 64 * 64; idx += TPB) {
            int nn = idx >> 6, kk = idx & 63;
            float v = (nn < 63 && kk < 63) ? W1p[kk * 63 + nn] : 0.0f;
            w1L[nn * FS + kk] = (_Float16)v;
        }
        if (tid < 64) b1L[tid] = (tid < 63) ? b1p[tid] : 0.0f;
        if (pass == 0) {
            if (tid < 64) w2L[tid] = (tid < 63) ? Wd2[tid] : 0.0f;
        } else {
            for (int idx = tid; idx < 64 * 3; idx += TPB) {
                int kk = idx / 3, jj = idx % 3;
                w2L[idx] = (kk < 63) ? Wc2[kk * 3 + jj] : 0.0f;
            }
        }
        __syncthreads();

        // -------- layer 1 via WMMA f16 -> f32 : hid = relu(f @ W1 + b1) --------
        const int asel = lane >> 4;          // 0: K 0-7/16-23 ; 1: K 8-15/24-31
        const int lc   = lane & 15;
#pragma unroll
        for (int mt = 0; mt < 2; ++mt) {
            const int mbase = wave * 32 + mt * 16;
            const int arow  = mbase + lc;

            // A fragments (ISA 16-bit A 16x32 layout), K-chunks 0 and 1
            v16h a0, a1;
            {
                const _Float16* ar = &featL[arow * FS];
                v8h lo0 = *(const v8h*)(ar + asel * 8);
                v8h hi0 = *(const v8h*)(ar + asel * 8 + 16);
                v8h lo1 = *(const v8h*)(ar + 32 + asel * 8);
                v8h hi1 = *(const v8h*)(ar + 32 + asel * 8 + 16);
#pragma unroll
                for (int i = 0; i < 8; ++i) {
                    a0[i] = lo0[i]; a0[8 + i] = hi0[i];
                    a1[i] = lo1[i]; a1[8 + i] = hi1[i];
                }
            }

#pragma unroll
            for (int nt = 0; nt < 4; ++nt) {
                const int ncol = nt * 16 + lc;
                const int kb   = asel * 8;    // lanes 0-15: K 0-7/16-23 ; lanes 16-31: K 8-15/24-31
                const _Float16* wr = &w1L[ncol * FS];
                v8h b0a = *(const v8h*)(wr + kb);
                v8h b0b = *(const v8h*)(wr + kb + 16);
                v8h b1a = *(const v8h*)(wr + 32 + kb);
                v8h b1b = *(const v8h*)(wr + 32 + kb + 16);
                v16h bf0, bf1;
#pragma unroll
                for (int i = 0; i < 8; ++i) {
                    bf0[i] = b0a[i]; bf0[8 + i] = b0b[i];
                    bf1[i] = b1a[i]; bf1[8 + i] = b1b[i];
                }

                v8f acc;
#pragma unroll
                for (int i = 0; i < 8; ++i) acc[i] = 0.0f;
                acc = __builtin_amdgcn_wmma_f32_16x16x32_f16(false, a0, false, bf0,
                                                             (short)0, acc, false, false);
                acc = __builtin_amdgcn_wmma_f32_16x16x32_f16(false, a1, false, bf1,
                                                             (short)0, acc, false, false);

                // bias + relu, scatter per C/D layout: VGPR r -> M = r (+8 for upper lanes)
                const float bias  = b1L[ncol];
                const int   rbase = mbase + (asel << 3);
#pragma unroll
                for (int r = 0; r < 8; ++r) {
                    float h = acc[r] + bias;
                    h = h > 0.0f ? h : 0.0f;
                    hidL[(rbase + r) * FS + ncol] = (_Float16)h;
                }
            }
        }
        __syncthreads();

        // -------- layer 2: thin GEMV per lane --------
        if (g < n) {
            const _Float16* h = &hidL[tid * FS];
            if (pass == 0) {
                float s = bd2[0];
#pragma unroll 9
                for (int k = 0; k < 63; ++k) s = fmaf((float)h[k], w2L[k], s);
                out[g] = s;
            } else {
                float s0 = bc2[0], s1 = bc2[1], s2 = bc2[2];
#pragma unroll 9
                for (int k = 0; k < 63; ++k) {
                    float hv = (float)h[k];
                    s0 = fmaf(hv, w2L[k * 3 + 0], s0);
                    s1 = fmaf(hv, w2L[k * 3 + 1], s1);
                    s2 = fmaf(hv, w2L[k * 3 + 2], s2);
                }
                out[n + 3 * g + 0] = s0;
                out[n + 3 * g + 1] = s1;
                out[n + 3 * g + 2] = s2;
            }
        }
        __syncthreads();
    }
}

extern "C" void kernel_launch(void* const* d_in, const int* in_sizes, int n_in,
                              void* d_out, int out_size, void* d_ws, size_t ws_size,
                              hipStream_t stream) {
    const float* x   = (const float*)d_in[0];
    const float* e0  = (const float*)d_in[1];
    const float* e1  = (const float*)d_in[2];
    const float* e2  = (const float*)d_in[3];
    const float* e3  = (const float*)d_in[4];
    const float* e4  = (const float*)d_in[5];
    const float* Wd1 = (const float*)d_in[6];
    const float* bd1 = (const float*)d_in[7];
    const float* Wd2 = (const float*)d_in[8];
    const float* bd2 = (const float*)d_in[9];
    const float* Wc1 = (const float*)d_in[10];
    const float* bc1 = (const float*)d_in[11];
    const float* Wc2 = (const float*)d_in[12];
    const float* bc2 = (const float*)d_in[13];
    float* out = (float*)d_out;

    const int n = in_sizes[0] / 3;
    const int blocks = (n + TPB - 1) / TPB;
    nerf_hashmlp_kernel<<<blocks, TPB, 0, stream>>>(x, e0, e1, e2, e3, e4,
                                                    Wd1, bd1, Wd2, bd2,
                                                    Wc1, bc1, Wc2, bc2, out, n);
}
